// AffineCouplingLayer_59433757442152
// MI455X (gfx1250) — matrix operand order repaired and checked by reading, not compiled, not run on hardware
//
#include <hip/hip_runtime.h>

typedef __attribute__((ext_vector_type(16))) _Float16 v16h;
typedef __attribute__((ext_vector_type(8)))  float    v8f;
typedef __attribute__((ext_vector_type(4)))  unsigned int v4u;

union V16H {
  v16h     h;
  v4u      q[2];
  _Float16 e[16];
};

#define B_ROWS   524288
#define NT_TILES (B_ROWS / 16)   // 32768
#define GRID     2048            // 2048*8 waves * 2 iters = 32768 tiles exactly

__device__ __forceinline__ v8f wmma_f16(v16h a, v16h b, v8f c) {
  // D(f32 16x16) = A(f16 16x32) * B(f16 32x16) + C
  return __builtin_amdgcn_wmma_f32_16x16x32_f16(false, a, false, b, (short)0, c,
                                                false, false);
}

__device__ __forceinline__ float relu(float v) {
  // select form: single v_max/cndmask, no canonicalize double-max
  return v > 0.0f ? v : 0.0f;
}

__device__ __forceinline__ float fast_tanh(float v) {
#if __has_builtin(__builtin_amdgcn_tanhf)
  return __builtin_amdgcn_tanhf(v);   // V_TANH_F32 (CDNA5 trans op)
#else
  return tanhf(v);
#endif
}

// Convert 16 f32 (K-contiguous in LDS) into one A-layout f16 group of 8+8.
__device__ __forceinline__ void lds_to_a16(const float* p, V16H& a) {
  // p points at K-base for this lane's low 8 halves; p+16 for high 8.
  const float4* P = (const float4*)p;
  float4 g;
  g = P[0]; a.e[0]  = (_Float16)g.x; a.e[1]  = (_Float16)g.y;
            a.e[2]  = (_Float16)g.z; a.e[3]  = (_Float16)g.w;
  g = P[1]; a.e[4]  = (_Float16)g.x; a.e[5]  = (_Float16)g.y;
            a.e[6]  = (_Float16)g.z; a.e[7]  = (_Float16)g.w;
  g = P[4]; a.e[8]  = (_Float16)g.x; a.e[9]  = (_Float16)g.y;
            a.e[10] = (_Float16)g.z; a.e[11] = (_Float16)g.w;
  g = P[5]; a.e[12] = (_Float16)g.x; a.e[13] = (_Float16)g.y;
            a.e[14] = (_Float16)g.z; a.e[15] = (_Float16)g.w;
}

__global__ __launch_bounds__(256) void affine_coupling_kernel(
    const float* __restrict__ x,
    const float* __restrict__ W1, const float* __restrict__ b1,
    const float* __restrict__ W2, const float* __restrict__ b2,
    const float* __restrict__ W3, const float* __restrict__ b3,
    float* __restrict__ outy, float* __restrict__ outs) {
  // Weights as f16, pre-swizzled into WMMA B-layout tiles (32x16 each).
  // Tile flat layout: [half_group 0|1][lane][8 halfs] -> 2x ds_load_b128/lane.
  __shared__ __align__(16) _Float16 wt[20 * 512];
  __shared__ float bia[3 * 64];
  // Per-wave 16x64 *f32* activation buffer, rows padded to 68 floats (272B):
  // row stride == 4 banks (mod 64) -> lanes 0..15 hit disjoint bank quads.
  __shared__ __align__(16) float hbuf[8][16 * 68];

  const int tid = threadIdx.x;

  // ---- one-time weight staging: f32 global -> f16 LDS in B-tile layout ----
  for (int idx = tid; idx < 20 * 512; idx += 256) {
    int t  = idx >> 9;        // tile 0..19
    int e  = idx & 511;
    int hg = e >> 8;          // half-group (K 0..15 vs 16..31 within lane-half)
    int ln = (e >> 3) & 31;   // lane
    int h8 = e & 7;
    int kb = (ln < 16) ? 0 : 8;
    int K  = hg * 16 + kb + h8;   // 0..31 within 32-row tile
    int n  = ln & 15;
    const float* W; int kc, nt;
    if (t < 4)       { W = W1; kc = 0;             nt = t; }
    else if (t < 12) { W = W2; kc = (t - 4) >> 2;  nt = (t - 4) & 3; }
    else             { W = W3; kc = (t - 12) >> 2; nt = (t - 12) & 3; }
    wt[idx] = (_Float16)W[(kc * 32 + K) * 64 + nt * 16 + n];
  }
  if (tid < 64) {
    bia[tid]       = b1[tid];
    bia[64 + tid]  = b2[tid];
    bia[128 + tid] = b3[tid];
  }
  __syncthreads();

  const int  lane  = tid & 31;
  const int  wave  = tid >> 5;
  const int  lh    = lane & 15;
  const bool hiL   = lane >= 16;
  const int  mbase = hiL ? 8 : 0;   // C-layout row offset for this lane half
  float*     hb    = hbuf[wave];
  // A-layout reload base (float index): row lh, K offset kb (0 or 8)
  const int  abase = lh * 68 + (hiL ? 8 : 0);

  for (int tile = blockIdx.x * 8 + wave; tile < NT_TILES; tile += GRID * 8) {
    const int row0 = tile << 4;

    // ---- A0: even columns of x, converted to f16 in WMMA A-layout ----
    V16H a0;
    {
      const float4* p =
          (const float4*)(x + (size_t)(row0 + lh) * 64 + (hiL ? 16 : 0));
      float4 f;
      f = p[0];  a0.e[0]  = (_Float16)f.x; a0.e[1]  = (_Float16)f.z;
      f = p[1];  a0.e[2]  = (_Float16)f.x; a0.e[3]  = (_Float16)f.z;
      f = p[2];  a0.e[4]  = (_Float16)f.x; a0.e[5]  = (_Float16)f.z;
      f = p[3];  a0.e[6]  = (_Float16)f.x; a0.e[7]  = (_Float16)f.z;
      f = p[8];  a0.e[8]  = (_Float16)f.x; a0.e[9]  = (_Float16)f.z;
      f = p[9];  a0.e[10] = (_Float16)f.x; a0.e[11] = (_Float16)f.z;
      f = p[10]; a0.e[12] = (_Float16)f.x; a0.e[13] = (_Float16)f.z;
      f = p[11]; a0.e[14] = (_Float16)f.x; a0.e[15] = (_Float16)f.z;
    }

    // ---- layer 1: h1 = relu(cin @ W1 + b1), 4 N-tiles, K=32 ----
    #pragma unroll
    for (int nt = 0; nt < 4; ++nt) {
      float bv = bia[nt * 16 + lh];
      v8f c;
      #pragma unroll
      for (int i = 0; i < 8; ++i) c[i] = bv;   // bias broadcast as C init
      V16H b;
      b.q[0] = *(const v4u*)&wt[nt * 512 + lane * 8];
      b.q[1] = *(const v4u*)&wt[nt * 512 + 256 + lane * 8];
      c = wmma_f16(a0.h, b.h, c);
      #pragma unroll
      for (int r = 0; r < 8; ++r)
        hb[(mbase + r) * 68 + nt * 16 + lh] = relu(c[r]);
    }
    asm volatile("s_wait_dscnt 0" ::: "memory");

    // ---- reload h1 as A-layout (two K-chunks of 32), cvt_pk on load ----
    V16H a20, a21;
    lds_to_a16(hb + abase,      a20);   // K 0..31
    lds_to_a16(hb + abase + 32, a21);   // K 32..63

    // ---- layer 2: h2 = relu(h1 @ W2 + b2) ----
    #pragma unroll
    for (int nt = 0; nt < 4; ++nt) {
      float bv = bia[64 + nt * 16 + lh];
      v8f c;
      #pragma unroll
      for (int i = 0; i < 8; ++i) c[i] = bv;
      V16H b0, b1v;
      b0.q[0]  = *(const v4u*)&wt[(4 + nt) * 512 + lane * 8];
      b0.q[1]  = *(const v4u*)&wt[(4 + nt) * 512 + 256 + lane * 8];
      b1v.q[0] = *(const v4u*)&wt[(8 + nt) * 512 + lane * 8];
      b1v.q[1] = *(const v4u*)&wt[(8 + nt) * 512 + 256 + lane * 8];
      c = wmma_f16(a20.h, b0.h, c);
      c = wmma_f16(a21.h, b1v.h, c);
      #pragma unroll
      for (int r = 0; r < 8; ++r)
        hb[(mbase + r) * 68 + nt * 16 + lh] = relu(c[r]);
    }
    asm volatile("s_wait_dscnt 0" ::: "memory");

    V16H a30, a31;
    lds_to_a16(hb + abase,      a30);
    lds_to_a16(hb + abase + 32, a31);

    // ---- layer 3: st = h2 @ W3 + b3 (no relu). tiles 0,1 = s; 2,3 = t ----
    v8f st[4];
    #pragma unroll
    for (int nt = 0; nt < 4; ++nt) {
      float bv = bia[128 + nt * 16 + lh];
      v8f c;
      #pragma unroll
      for (int i = 0; i < 8; ++i) c[i] = bv;
      V16H b0, b1v;
      b0.q[0]  = *(const v4u*)&wt[(12 + nt) * 512 + lane * 8];
      b0.q[1]  = *(const v4u*)&wt[(12 + nt) * 512 + 256 + lane * 8];
      b1v.q[0] = *(const v4u*)&wt[(16 + nt) * 512 + lane * 8];
      b1v.q[1] = *(const v4u*)&wt[(16 + nt) * 512 + 256 + lane * 8];
      c = wmma_f16(a30.h, b0.h, c);
      c = wmma_f16(a31.h, b1v.h, c);
      st[nt] = c;
    }

    // ---- epilogue: s = tanh(.), y_odd = x_odd*exp(s)+t, y_even = x_even ----
    // C layout => lanes 0..15 share row (mbase+r), n consecutive: the
    // (2j,2j+1) pair loads/stores are fully coalesced 8B accesses.
    #pragma unroll
    for (int jt = 0; jt < 2; ++jt) {
      v8f sa = st[jt];
      v8f ta = st[2 + jt];
      const int j = jt * 16 + lh;          // odd-column index 0..31
      #pragma unroll
      for (int r = 0; r < 8; ++r) {
        const int row = row0 + mbase + r;
        float sv = fast_tanh(sa[r]);
        float es = __expf(sv);
        const float2 xv = *(const float2*)(x + (size_t)row * 64 + 2 * j);
        float2 yv;
        yv.x = xv.x;                        // even col passes through
        yv.y = xv.y * es + ta[r];           // odd col transformed
        *(float2*)(outy + (size_t)row * 64 + 2 * j) = yv;
        outs[(size_t)row * 32 + j] = sv;
      }
    }
  }
}

extern "C" void kernel_launch(void* const* d_in, const int* in_sizes, int n_in,
                              void* d_out, int out_size, void* d_ws,
                              size_t ws_size, hipStream_t stream) {
  const float* x  = (const float*)d_in[0];
  const float* W1 = (const float*)d_in[1];
  const float* b1 = (const float*)d_in[2];
  const float* W2 = (const float*)d_in[3];
  const float* b2 = (const float*)d_in[4];
  const float* W3 = (const float*)d_in[5];
  const float* b3 = (const float*)d_in[6];
  float* outy = (float*)d_out;
  float* outs = outy + (size_t)B_ROWS * 64;   // tuple (y, s) flattened

  affine_coupling_kernel<<<dim3(GRID), dim3(256), 0, stream>>>(
      x, W1, b1, W2, b2, W3, b3, outy, outs);
}
